// MultiHeadedAttention_14516989460888
// MI455X (gfx1250) — matrix operand order repaired and checked by reading, not compiled, not run on hardware
//
#include <hip/hip_runtime.h>
#include <math.h>

typedef __attribute__((ext_vector_type(16))) _Float16 v16h;
typedef __attribute__((ext_vector_type(8)))  _Float16 v8h;
typedef __attribute__((ext_vector_type(8)))  float    v8f;

#define S_LEN 2048
#define DMODEL 128
#define NHEAD 8
#define HDK 16
#define NBATCH 4
#define QBLK 4
#define LOG2E 1.44269504088896340736f

#if __has_builtin(__builtin_amdgcn_exp2f)
#define EXP2(x) __builtin_amdgcn_exp2f(x)
#else
#define EXP2(x) exp2f(x)
#endif

// ---------- fragment helpers ----------

// convert 8 contiguous f32 -> v8h
__device__ __forceinline__ v8h load8_f32_h(const float* p) {
  const float4* q = (const float4*)p;
  float4 u = q[0], v = q[1];
  v8h r;
  r[0] = (_Float16)u.x; r[1] = (_Float16)u.y; r[2] = (_Float16)u.z; r[3] = (_Float16)u.w;
  r[4] = (_Float16)v.x; r[5] = (_Float16)v.y; r[6] = (_Float16)v.z; r[7] = (_Float16)v.w;
  return r;
}

__device__ __forceinline__ v16h pack16(v8h lo, v8h hi) {
  v16h a;
#pragma unroll
  for (int i = 0; i < 8; ++i) { a[i] = lo[i]; a[i + 8] = hi[i]; }
  return a;
}

// A-fragment (16x32 f16): lane row m=lane&15, K = sel*16+e (16 contiguous)
__device__ __forceinline__ v16h afrag_f32(const float* row, int k0, int sel) {
  return pack16(load8_f32_h(row + k0 + sel * 16),
                load8_f32_h(row + k0 + sel * 16 + 8));
}
__device__ __forceinline__ v16h afrag_f16(const _Float16* row, int sel) {
  v8h lo = *(const v8h*)(row + sel * 16);
  v8h hi = *(const v8h*)(row + sel * 16 + 8);
  return pack16(lo, hi);
}

// B-fragment (32x16 f16): lane col n=lane&15, K = sel*16+e (16 contiguous)
__device__ __forceinline__ v16h bfrag_f32(const float* row, int k0, int sel) {
  return pack16(load8_f32_h(row + k0 + sel * 16),
                load8_f32_h(row + k0 + sel * 16 + 8));
}

__device__ __forceinline__ v8f wmma_f16(v16h a, v16h b, v8f c) {
  return __builtin_amdgcn_wmma_f32_16x16x32_f16(false, a, false, b, (short)0, c,
                                                false, false);
}

// ---------- stage 1: fused projections ----------
// grid.x = 512 row tiles (16 rows of B*S), grid.y = mode (0:Q 1:posQ 2:K 3:posK 4:V)
// modes 0-3: compute D = W_tile x X^T -> lane holds fixed row s, contiguous d
// mode 4:    compute D = X x W^T      -> vector store into transposed Vt
__global__ __launch_bounds__(128) void proj_kernel(
    const float* __restrict__ query, const float* __restrict__ key,
    const float* __restrict__ value, const float* __restrict__ pos,
    const float* __restrict__ W0, const float* __restrict__ b0,
    const float* __restrict__ W1, const float* __restrict__ b1,
    const float* __restrict__ W2, const float* __restrict__ b2,
    _Float16* __restrict__ Qc, _Float16* __restrict__ Kc,
    _Float16* __restrict__ Vt) {
  const int g    = blockIdx.x;
  const int mode = blockIdx.y;
  const int wave = threadIdx.x >> 5;
  const int lane = threadIdx.x & 31;
  const int sel  = lane >> 4;
  const int ln   = lane & 15;

  const float* X; const float* W; const float* bias; float scale;
  if      (mode == 0) { X = query; W = W0; bias = b0; scale = 0.25f * LOG2E; }
  else if (mode == 1) { X = pos;   W = W0; bias = b0; scale = LOG2E;         }
  else if (mode == 2) { X = key;   W = W1; bias = b1; scale = 1.0f;          }
  else if (mode == 3) { X = pos;   W = W1; bias = b1; scale = 1.0f;          }
  else                { X = value; W = W2; bias = b2; scale = 1.0f;          }

  const long rg0 = (long)g * 16;

  if (mode < 4) {
    // lane's sequence row (fixed): B-fragment of X^T
    const long row = rg0 + ln;
    const int  bb  = (int)(row >> 11);
    const int  ss  = (int)(row & 2047);
    const float* xr = X + row * DMODEL;
    v16h xb0 = bfrag_f32(xr,  0, sel);
    v16h xb1 = bfrag_f32(xr, 32, sel);
    v16h xb2 = bfrag_f32(xr, 64, sel);
    v16h xb3 = bfrag_f32(xr, 96, sel);

    _Float16* base = (mode < 2) ? Qc : Kc;
    const int off  = (mode & 1) ? HDK : 0;

#pragma unroll
    for (int t = 0; t < 2; ++t) {
      const int nt = wave + 4 * t;              // head index (tile of 16 cols)
      const float* wr = W + (long)(nt * 16 + ln) * DMODEL;  // A rows of W
      v8f acc = {};
      acc = wmma_f16(afrag_f32(wr,  0, sel), xb0, acc);
      acc = wmma_f16(afrag_f32(wr, 32, sel), xb1, acc);
      acc = wmma_f16(afrag_f32(wr, 64, sel), xb2, acc);
      acc = wmma_f16(afrag_f32(wr, 96, sel), xb3, acc);

      const float* bp = bias + nt * 16 + 8 * sel;   // uniform across lanes
      v8h o;
#pragma unroll
      for (int r = 0; r < 8; ++r) o[r] = (_Float16)((acc[r] + bp[r]) * scale);
      _Float16* dst =
          base + (((long)(bb * NHEAD + nt) * S_LEN + ss) << 5) + off + 8 * sel;
      *(v8h*)dst = o;
    }
  } else {
    // V: D = X x W^T, lane holds fixed column d, 8 contiguous s rows
    const float* ar = X + (rg0 + ln) * DMODEL;
    v16h a0 = afrag_f32(ar,  0, sel);
    v16h a1 = afrag_f32(ar, 32, sel);
    v16h a2 = afrag_f32(ar, 64, sel);
    v16h a3 = afrag_f32(ar, 96, sel);

#pragma unroll
    for (int t = 0; t < 2; ++t) {
      const int nt = wave + 4 * t;
      const int n  = nt * 16 + ln;
      const float* wr = W + (long)n * DMODEL;
      v8f acc = {};
      acc = wmma_f16(a0, bfrag_f32(wr,  0, sel), acc);
      acc = wmma_f16(a1, bfrag_f32(wr, 32, sel), acc);
      acc = wmma_f16(a2, bfrag_f32(wr, 64, sel), acc);
      acc = wmma_f16(a3, bfrag_f32(wr, 96, sel), acc);

      const float bn = bias[n];
      const int hq = n >> 4;
      const int d  = n & 15;
      const long row = rg0 + 8 * sel;
      const int bb = (int)(row >> 11);
      const int s0 = (int)(row & 2047);
      _Float16* dst =
          Vt + ((((long)(bb * NHEAD + hq)) * HDK + d) * S_LEN + s0);
      v8h o;
#pragma unroll
      for (int r = 0; r < 8; ++r) o[r] = (_Float16)(acc[r] + bn);
      *(v8h*)dst = o;
    }
  }
}

// ---------- stage 2: flash attention ----------
// 4 waves per block, QBLK=4 query tiles per wave: K/V fragments for each
// 32-key window are loaded once and reused across 4 independent
// score->softmax->PV chains (4x less L2 traffic, hazard-hiding ILP).
// Qc holds q*0.25*log2e and pos_q*log2e -> scores already in log2 domain.
__global__ __launch_bounds__(128) void attn_kernel(
    const _Float16* __restrict__ Qc, const _Float16* __restrict__ Kc,
    const _Float16* __restrict__ Vt, _Float16* __restrict__ Xo) {
  const int wave = threadIdx.x >> 5;
  const int lane = threadIdx.x & 31;
  const int sel  = lane >> 4;
  const int ln   = lane & 15;
  const int bh   = blockIdx.y;           // 0..31
  const int b    = bh >> 3;
  const int h    = bh & 7;
  const int qt0  = (blockIdx.x * 4 + wave) * QBLK;   // blockIdx.x: 0..7

  const _Float16* Qb = Qc + ((long)bh * S_LEN << 5);
  const _Float16* Kb = Kc + ((long)bh * S_LEN << 5);
  const _Float16* Vr = Vt + (long)bh * HDK * S_LEN + (long)ln * S_LEN;

  v16h  qb[QBLK];
  float M[QBLK], L[QBLK];
  v8f   acc[QBLK];
#pragma unroll
  for (int j = 0; j < QBLK; ++j) {
    const int sq = (qt0 + j) * 16 + ln;
    qb[j]  = *(const v16h*)(Qb + ((long)sq << 5) + sel * 16);
    M[j]   = -INFINITY;
    L[j]   = 0.0f;
    acc[j] = (v8f){};
  }

  for (int i = 0; i < S_LEN / 32; ++i) {
    const int kb = i * 32;
    // shared per-window fragments
    v16h ka0 = afrag_f16(Kb + ((long)(kb + ln) << 5), sel);
    v16h ka1 = afrag_f16(Kb + ((long)(kb + 16 + ln) << 5), sel);
    v16h va  = afrag_f16(Vr + kb, sel);

#pragma unroll
    for (int j = 0; j < QBLK; ++j) {
      // S^T = K_tile x Q^T : lane col = query, rows = keys r+8*sel
      v8f z0 = {}, z1 = {};
      v8f st0 = wmma_f16(ka0, qb[j], z0);
      v8f st1 = wmma_f16(ka1, qb[j], z1);

      // --- online softmax in log2 domain (per query column) ---
      float mloc = st0[0];
#pragma unroll
      for (int r = 1; r < 8; ++r) mloc = fmaxf(mloc, st0[r]);
#pragma unroll
      for (int r = 0; r < 8; ++r) mloc = fmaxf(mloc, st1[r]);
      mloc = fmaxf(mloc, __shfl_xor(mloc, 16, 32));
      const float Mn    = fmaxf(M[j], mloc);
      const float alpha = EXP2(M[j] - Mn);
      M[j] = Mn;

      float p0[8], p1[8], lsum = 0.0f;
#pragma unroll
      for (int r = 0; r < 8; ++r) {
        p0[r] = EXP2(st0[r] - Mn);
        p1[r] = EXP2(st1[r] - Mn);
        lsum += p0[r] + p1[r];
      }
      lsum += __shfl_xor(lsum, 16, 32);
      L[j] = L[j] * alpha + lsum;

      // build B-fragment of P^T (32 keys x 16 queries) via xor-16 shuffles
      v16h pb;
#pragma unroll
      for (int e = 0; e < 8; ++e) {
        const float o0 = __shfl_xor(p0[e], 16, 32);
        const float o1 = __shfl_xor(p1[e], 16, 32);
        pb[e]     = (_Float16)(sel ? o1 : p0[e]);
        pb[e + 8] = (_Float16)(sel ? p1[e] : o0);
      }

#pragma unroll
      for (int r = 0; r < 8; ++r) acc[j][r] *= alpha;
      acc[j] = wmma_f16(va, pb, acc[j]);   // X^T += V^T * P^T
    }
  }

#pragma unroll
  for (int j = 0; j < QBLK; ++j) {
    const float invL = 1.0f / L[j];
    const int sq = (qt0 + j) * 16 + ln;
    v8h o;
#pragma unroll
    for (int r = 0; r < 8; ++r) o[r] = (_Float16)(acc[j][r] * invL);
    _Float16* dst =
        Xo + ((long)(b * S_LEN + sq)) * DMODEL + h * HDK + sel * 8;
    *(v8h*)dst = o;
  }
}

// ---------- stage 3: output projection  D = Wo x Xo^T  ----------
// lane holds fixed row s, 8 contiguous output columns -> vector f32 stores
__global__ __launch_bounds__(128) void outproj_kernel(
    const _Float16* __restrict__ Xo, const float* __restrict__ Wo,
    const float* __restrict__ bo, float* __restrict__ out) {
  const int g    = blockIdx.x;
  const int wave = threadIdx.x >> 5;
  const int lane = threadIdx.x & 31;
  const int sel  = lane >> 4;
  const int ln   = lane & 15;

  const long row = (long)g * 16 + ln;          // lane's sequence row (fixed)
  const _Float16* xr = Xo + row * DMODEL;
  v16h xb0 = *(const v16h*)(xr +  0 + sel * 16);
  v16h xb1 = *(const v16h*)(xr + 32 + sel * 16);
  v16h xb2 = *(const v16h*)(xr + 64 + sel * 16);
  v16h xb3 = *(const v16h*)(xr + 96 + sel * 16);

#pragma unroll
  for (int t = 0; t < 2; ++t) {
    const int nt = wave + 4 * t;
    const float* wr = Wo + (long)(nt * 16 + ln) * DMODEL;
    v8f acc = {};
    acc = wmma_f16(afrag_f32(wr,  0, sel), xb0, acc);
    acc = wmma_f16(afrag_f32(wr, 32, sel), xb1, acc);
    acc = wmma_f16(afrag_f32(wr, 64, sel), xb2, acc);
    acc = wmma_f16(afrag_f32(wr, 96, sel), xb3, acc);

    const float* bp = bo + nt * 16 + 8 * sel;  // uniform across lanes
    float4 o0, o1;
    o0.x = acc[0] + bp[0]; o0.y = acc[1] + bp[1];
    o0.z = acc[2] + bp[2]; o0.w = acc[3] + bp[3];
    o1.x = acc[4] + bp[4]; o1.y = acc[5] + bp[5];
    o1.z = acc[6] + bp[6]; o1.w = acc[7] + bp[7];
    float* dst = out + row * DMODEL + nt * 16 + 8 * sel;
    *(float4*)dst       = o0;
    *(float4*)(dst + 4) = o1;
  }
}

// ---------- launch ----------
extern "C" void kernel_launch(void* const* d_in, const int* in_sizes, int n_in,
                              void* d_out, int out_size, void* d_ws,
                              size_t ws_size, hipStream_t stream) {
  const float* query = (const float*)d_in[0];
  const float* key_  = (const float*)d_in[1];
  const float* value = (const float*)d_in[2];
  const float* pos   = (const float*)d_in[3];
  const float* W0 = (const float*)d_in[4];
  const float* b0 = (const float*)d_in[5];
  const float* W1 = (const float*)d_in[6];
  const float* b1 = (const float*)d_in[7];
  const float* W2 = (const float*)d_in[8];
  const float* b2 = (const float*)d_in[9];
  const float* Wo = (const float*)d_in[10];
  const float* bo = (const float*)d_in[11];

  const size_t nQK = (size_t)NBATCH * NHEAD * S_LEN * 32;    // 2M f16 each
  const size_t nV  = (size_t)NBATCH * NHEAD * HDK * S_LEN;   // 1M f16
  _Float16* Qc = (_Float16*)d_ws;
  _Float16* Kc = Qc + nQK;
  _Float16* Vt = Kc + nQK;
  _Float16* Xo = Vt + nV;

  proj_kernel<<<dim3(512, 5), 128, 0, stream>>>(query, key_, value, pos, W0,
                                                b0, W1, b1, W2, b2, Qc, Kc, Vt);
  attn_kernel<<<dim3(8, 32), 128, 0, stream>>>(Qc, Kc, Vt, Xo);
  outproj_kernel<<<dim3(512), 128, 0, stream>>>(Xo, Wo, bo, (float*)d_out);
}
